// ShiftMLP_60601988546953
// MI455X (gfx1250) — matrix operand order repaired and checked by reading, not compiled, not run on hardware
//
#include <hip/hip_runtime.h>

#define DIM 256
#define HH  128
#define WWID 128
#define BB  8
#define NN  (HH * WWID)
#define ASTR 264   // LDS A-tile row stride in halves (256 + 8 pad)

typedef __attribute__((ext_vector_type(16))) _Float16 v16h;
typedef __attribute__((ext_vector_type(8)))  float    v8f;

union FragAB { v16h v; float4 f4[2]; _Float16 h[16]; };
union FragC  { v8f  v; float f[8]; };

// ---------------------------------------------------------------------------
// Pack W1 / W2 (f32 row-major [K=256][N=256]) into WMMA-B fragment-major f16:
// packed half index e = kt*8192 + nt*512 + lane*16 + h
//   K = kt*32 + (h<8 ? 8*(lane>>4)+h : 16+8*(lane>>4)+(h-8)),  N = nt*16 + (lane&15)
// ---------------------------------------------------------------------------
__global__ __launch_bounds__(256) void pack_weights(
    const float* __restrict__ W1, const float* __restrict__ W2,
    _Float16* __restrict__ w1p, _Float16* __restrict__ w2p) {
  int idx  = blockIdx.x * 256 + threadIdx.x;   // 0 .. 131071
  int mat  = idx >> 16;
  int e    = idx & 65535;
  int kt   = e >> 13;
  int nt   = (e >> 9) & 15;
  int lane = (e >> 4) & 31;
  int h    = e & 15;
  int K = kt * 32 + ((h < 8) ? 8 * (lane >> 4) + h : 16 + 8 * (lane >> 4) + (h - 8));
  int N = (nt << 4) + (lane & 15);
  const float* src = mat ? W2 : W1;
  _Float16*    dst = mat ? w2p : w1p;
  dst[e] = (_Float16)src[K * DIM + N];
}

// ---------------------------------------------------------------------------
// 64x256 GEMM tile: 8 waves; wave w -> M-block (w&3)*16, N-half (w>>2).
// Per kt: 1 A-frag (2x ds_load_b128) + 8 B-frags batched (16x global_load_b128
// in one clause) + 8 independent WMMAs; scheduler overlaps next kt's loads.
// ---------------------------------------------------------------------------
__device__ __forceinline__ void gemm_tile(const _Float16* at,
                                          const _Float16* __restrict__ wp,
                                          int lane, int mb, int nh,
                                          FragC acc[8]) {
  const int laneHi = lane >> 4;
  const int arow   = mb * 16 + (lane & 15);
  const _Float16* bbase = wp + (size_t)(nh * 8) * 512 + lane * 16;
#pragma unroll
  for (int q = 0; q < 8; ++q)
#pragma unroll
    for (int e2 = 0; e2 < 8; ++e2) acc[q].f[e2] = 0.0f;

#pragma unroll
  for (int kt = 0; kt < 8; ++kt) {
    FragAB a;
    const _Float16* ap = at + arow * ASTR + kt * 32 + laneHi * 8;
    a.f4[0] = *(const float4*)(ap);        // K chunk base0 .. base0+7
    a.f4[1] = *(const float4*)(ap + 16);   // K chunk base0+16 .. base0+23
    FragAB bf[8];
#pragma unroll
    for (int q = 0; q < 8; ++q) {
      const _Float16* bq = bbase + kt * 8192 + q * 512;
      bf[q].f4[0] = *(const float4*)(bq);
      bf[q].f4[1] = *(const float4*)(bq + 8);
    }
#pragma unroll
    for (int q = 0; q < 8; ++q)
      acc[q].v = __builtin_amdgcn_wmma_f32_16x16x32_f16(
          false, a.v, false, bf[q].v, (short)0, acc[q].v, false, false);
  }
}

// ---------------------------------------------------------------------------
// K1: shift-H (per 64-ch group: src row = i+2-g, zero OOB) + GEMM1 + b1 -> y1 f16
// grid: 2048 = B * H * 2 (j0 in {0,64}); block computes 64 rows x 256 cols
// ---------------------------------------------------------------------------
__global__ __launch_bounds__(256) void k1_shiftH_gemm(
    const float* __restrict__ x, const _Float16* __restrict__ w1p,
    const float* __restrict__ b1, _Float16* __restrict__ y1) {
  __shared__ _Float16 at[64 * ASTR];
  const int bx = blockIdx.x;
  const int b  = bx >> 8;
  const int r  = bx & 255;
  const int i  = r >> 1;
  const int j0 = (r & 1) << 6;
  const int t  = threadIdx.x;

  const float* xb = x + (size_t)b * NN * DIM;
#pragma unroll
  for (int v = 0; v < 16; ++v) {
    int f  = v * 256 + t;        // float4 index within 64x256 tile
    int jl = f >> 6;             // tile row (j offset)
    int fi = f & 63;             // float4 within row -> k = fi*4
    int g  = fi >> 4;            // channel group of this k
    int isrc = i + 2 - g;        // shifted source row
    union { _Float16 h[4]; uint2 u; } tmp;
    if (isrc >= 0 && isrc < HH) {
      float4 s = *(const float4*)(xb + (size_t)(isrc * WWID + j0 + jl) * DIM + fi * 4);
      tmp.h[0] = (_Float16)s.x; tmp.h[1] = (_Float16)s.y;
      tmp.h[2] = (_Float16)s.z; tmp.h[3] = (_Float16)s.w;
    } else {
      tmp.u = make_uint2(0u, 0u);
    }
    *(uint2*)&at[jl * ASTR + fi * 4] = tmp.u;
  }
  __syncthreads();

  const int lane = t & 31, w = t >> 5;
  const int mb = w & 3, nh = w >> 2;
  const int laneHi = lane >> 4, lm = lane & 15;
  FragC acc[8];
  gemm_tile(at, w1p, lane, mb, nh, acc);

  _Float16* y1b = y1 + (size_t)b * NN * DIM + (size_t)(i * WWID) * DIM;
#pragma unroll
  for (int q = 0; q < 8; ++q) {
    int c = (nh * 8 + q) * 16 + lm;
    float bias = b1[c];
#pragma unroll
    for (int v = 0; v < 8; ++v) {
      int jl = mb * 16 + v + laneHi * 8;
      y1b[(size_t)(j0 + jl) * DIM + c] = (_Float16)(acc[q].f[v] + bias);
    }
  }
}

// ---------------------------------------------------------------------------
// K2: depthwise 3x3 SAME conv + dw_bias + ReLU, storing directly into the
// pre-shifted transposed tile layout consumed by GEMM2:
//   a2s[b][qt][d][jl]  (qt = q>>6, jl = q&63, q = q' + s_g, q' = p & 16383,
//                       d = p>>14, p = n*256 + c, s_g = (d>>6) - 2)
// Per thread (channel pair c,c+1 ; j-half jh): d = 2i + jh is constant, so
// jl/qt-offset/validity are loop constants. Sliding 3-column window:
// 3 packed u32 loads per output pair (y1 read ~1x).
// grid: 1024 = B*H; 256 threads = 128 channel-pairs x 2 j-halves
// ---------------------------------------------------------------------------
__global__ __launch_bounds__(256) void k2_dwconv_shiftpack(
    const _Float16* __restrict__ y1, const float* __restrict__ kern,
    const float* __restrict__ dwb, _Float16* __restrict__ a2s) {
  const int bx = blockIdx.x;
  const int b = bx >> 7, i = bx & 127;
  const int t = threadIdx.x;
  const int c  = (t & 127) * 2;        // channel pair c, c+1
  const int jh = t >> 7;               // j in [jh*64, jh*64+64)
  const int d  = 2 * i + jh;
  const int s  = (d >> 6) - 2;         // shift in {-2,-1,0,1}
  const int w0 = c & 127, w1 = (c + 1) & 127;
  const bool v0 = (unsigned)(w0 + s) < 128u;   // store-validity per channel
  const bool v1 = (unsigned)(w1 + s) < 128u;
  const int jl0 = (c + s) & 63,  jl1 = (c + 1 + s) & 63;
  const int qc0 = (c + s) >> 6,  qc1 = (c + 1 + s) >> 6;

  float k0[9], k1[9];
#pragma unroll
  for (int e = 0; e < 9; ++e) { k0[e] = kern[e * DIM + c]; k1[e] = kern[e * DIM + c + 1]; }
  const float bias0 = dwb[c], bias1 = dwb[c + 1];

  const _Float16* yb = y1 + (size_t)b * NN * DIM;
  _Float16* ab = a2s + (size_t)b * NN * DIM;

  float2 win[3][3];    // [col: j-1, j, j+1][row di]
  auto loadcol = [&](int jj, float2 out[3]) {
#pragma unroll
    for (int di = 0; di < 3; ++di) {
      int ii = i + di - 1;
      float2 rv; rv.x = 0.0f; rv.y = 0.0f;
      if (ii >= 0 && ii < HH && jj >= 0 && jj < WWID) {
        union { unsigned u; _Float16 h[2]; } tv;
        tv.u = *(const unsigned*)(yb + (size_t)(ii * WWID + jj) * DIM + c);
        rv.x = (float)tv.h[0]; rv.y = (float)tv.h[1];
      }
      out[di] = rv;
    }
  };

  const int jbase = jh * 64;
  loadcol(jbase - 1, win[0]);
  loadcol(jbase,     win[1]);
  for (int jj = 0; jj < 64; ++jj) {
    int j = jbase + jj;
    loadcol(j + 1, win[2]);
    float s0 = bias0, s1 = bias1;
#pragma unroll
    for (int dj = 0; dj < 3; ++dj)
#pragma unroll
      for (int di = 0; di < 3; ++di) {
        s0 += win[dj][di].x * k0[di * 3 + dj];
        s1 += win[dj][di].y * k1[di * 3 + dj];
      }
    _Float16 r0 = (_Float16)fmaxf(s0, 0.0f);
    _Float16 r1 = (_Float16)fmaxf(s1, 0.0f);
    int base = jj * 4 * 16384 + d * 64;        // (j&63)*4 tiles + d row
    if (v0) ab[base + qc0 * 16384 + jl0] = r0;
    if (v1) ab[base + qc1 * 16384 + jl1] = r1;
#pragma unroll
    for (int di = 0; di < 3; ++di) { win[0][di] = win[1][di]; win[1][di] = win[2][di]; }
  }
}

// ---------------------------------------------------------------------------
// K3: GEMM2 + b2 -> out (f32). A-tile is one contiguous 32KB block of a2s:
// 8x global_load_b128 per thread, LDS transpose via packed ds_store_b32.
// Entries whose shift wrapped the 128-field were never stored by K2 and are
// zero-selected here with the complementary predicate (ww - s) in [0,128).
// grid: 2048 = B * 256 q-tiles of 64
// ---------------------------------------------------------------------------
__global__ __launch_bounds__(256) void k3_gemm2(
    const _Float16* __restrict__ a2s, const _Float16* __restrict__ w2p,
    const float* __restrict__ b2, float* __restrict__ out) {
  __shared__ _Float16 at[64 * ASTR];
  const int bx = blockIdx.x;
  const int b  = bx >> 8;
  const int qt = bx & 255;
  const int ww0 = (qt & 1) << 6;
  const int t  = threadIdx.x;

  const _Float16* src = a2s + (size_t)b * NN * DIM + (size_t)qt * 16384;
  const int d0 = (t & 127) * 2;       // d pair (same 64-group -> same shift)
  const int jh = (t >> 7) * 32;       // jl half
  const int s0 = (d0 >> 6) - 2;
  union { float4 f4[4]; _Float16 h[32]; } A0, A1;
#pragma unroll
  for (int v = 0; v < 4; ++v) {
    A0.f4[v] = *(const float4*)(src + d0 * 64 + jh + v * 8);
    A1.f4[v] = *(const float4*)(src + (d0 + 1) * 64 + jh + v * 8);
  }
#pragma unroll
  for (int v = 0; v < 32; ++v) {
    int jl = jh + v;
    bool valid = (unsigned)(ww0 + jl - s0) < 128u;
    union { unsigned u; _Float16 h[2]; } pk;
    pk.h[0] = valid ? A0.h[v] : (_Float16)0.0f;
    pk.h[1] = valid ? A1.h[v] : (_Float16)0.0f;
    *(unsigned*)&at[jl * ASTR + d0] = pk.u;
  }
  __syncthreads();

  const int lane = t & 31, w = t >> 5;
  const int mb = w & 3, nh = w >> 2;
  const int laneHi = lane >> 4, lm = lane & 15;
  FragC acc[8];
  gemm_tile(at, w2p, lane, mb, nh, acc);

  float* ob = out + (size_t)b * NN * DIM + (size_t)(qt << 6) * DIM;
#pragma unroll
  for (int q = 0; q < 8; ++q) {
    int e = (nh * 8 + q) * 16 + lm;
    float bias = b2[e];
#pragma unroll
    for (int v = 0; v < 8; ++v) {
      int jlo = mb * 16 + v + laneHi * 8;
      ob[(size_t)jlo * DIM + e] = acc[q].f[v] + bias;
    }
  }
}

// ---------------------------------------------------------------------------
extern "C" void kernel_launch(void* const* d_in, const int* in_sizes, int n_in,
                              void* d_out, int out_size, void* d_ws, size_t ws_size,
                              hipStream_t stream) {
  (void)in_sizes; (void)n_in; (void)out_size; (void)ws_size;
  const float* x   = (const float*)d_in[0];
  const float* W1  = (const float*)d_in[1];
  const float* b1  = (const float*)d_in[2];
  const float* dwk = (const float*)d_in[3];
  const float* dwb = (const float*)d_in[4];
  const float* W2  = (const float*)d_in[5];
  const float* b2  = (const float*)d_in[6];

  // workspace: [w1p 128KB][w2p 128KB][a2s 64MB]  (~67.4 MB total)
  _Float16* w1p = (_Float16*)d_ws;
  _Float16* w2p = (_Float16*)((char*)d_ws + 131072);
  _Float16* a2s = (_Float16*)((char*)d_ws + 262144);

  // y1 (f16, 67MB) staged in first half of d_out; K3 overwrites d_out last.
  _Float16* y1  = (_Float16*)d_out;
  float*    out = (float*)d_out;

  pack_weights       <<<512,  256, 0, stream>>>(W1, W2, w1p, w2p);
  k1_shiftH_gemm     <<<2048, 256, 0, stream>>>(x, w1p, b1, y1);
  k2_dwconv_shiftpack<<<1024, 256, 0, stream>>>(y1, dwk, dwb, a2s);
  k3_gemm2           <<<2048, 256, 0, stream>>>(a2s, w2p, b2, out);
}
